// SelfAttention_9629316677783
// MI455X (gfx1250) — matrix operand order repaired and checked by reading, not compiled, not run on hardware
//
#include <hip/hip_runtime.h>
#include <hip/hip_bf16.h>

typedef _Float16 f16;
typedef unsigned int u32;
typedef unsigned long long u64;
typedef __attribute__((ext_vector_type(16))) _Float16 v16h;
typedef __attribute__((ext_vector_type(8)))  _Float16 v8h;
typedef __attribute__((ext_vector_type(8)))  float    v8f;
typedef __attribute__((ext_vector_type(4)))  u32      u32x4;
typedef __attribute__((ext_vector_type(8)))  int      i32x8;
typedef __attribute__((ext_vector_type(4)))  int      i32x4;

#define DEV __device__ __forceinline__

// ---------------- constants -----------------------------------------------
#define BATCH   4
#define SEQ     2048
#define DIM     1024
#define NHEADS  16
#define HDIM    64
#define MTOT    (BATCH * SEQ)          // 8192 rows
#define KSTEP   32                     // K per TDM tile / per WMMA step (GEMM)
#define PITCH   80                     // LDS pitch for 32-elem rows (64B+16B pad)
#define PITCHK  144                    // LDS pitch for 64-elem rows (128B+16B pad)

// ---------------- WMMA fragment helpers (wave32 layouts, ISA 7.12.2) -------
// A 16x32 f16: lane<16 row M=lane, K={0..7,16..23}; lane>=16 row M=lane-16,
// K={8..15,24..31}.  p = row base (at k0); lh = lane>>4.
DEV v16h load_frag_a(const f16* p, int lh) {
    v8h a = *(const v8h*)(p + lh * 8);
    v8h b = *(const v8h*)(p + 16 + lh * 8);
    return __builtin_shufflevector(a, b, 0,1,2,3,4,5,6,7,8,9,10,11,12,13,14,15);
}
// B 32x16 f16: lane<16 col N=lane, K={0..7,16..23}; lane>=16 col N=lane-16,
// K={8..15,24..31}.  p = col base + k0 + (lh?8:0)
DEV v16h load_frag_b(const f16* p) {
    v8h a = *(const v8h*)(p);
    v8h b = *(const v8h*)(p + 16);
    return __builtin_shufflevector(a, b, 0,1,2,3,4,5,6,7,8,9,10,11,12,13,14,15);
}
// Same fragments out of a padded LDS tile; rowptr points at the 32-element
// K-group within the row (byte offsets below stay < 64).
DEV v16h lds_frag_a(const char* rowptr, int lh) {
    v8h a = *(const v8h*)(rowptr + lh * 16);
    v8h b = *(const v8h*)(rowptr + 32 + lh * 16);
    return __builtin_shufflevector(a, b, 0,1,2,3,4,5,6,7,8,9,10,11,12,13,14,15);
}
DEV v16h lds_frag_b(const char* rowptr, int lh) {
    v8h a = *(const v8h*)(rowptr + lh * 16);
    v8h b = *(const v8h*)(rowptr + 32 + lh * 16);
    return __builtin_shufflevector(a, b, 0,1,2,3,4,5,6,7,8,9,10,11,12,13,14,15);
}
DEV v8f wmma_f16(v16h a, v16h b, v8f c) {
    return __builtin_amdgcn_wmma_f32_16x16x32_f16(
        false, a, false, b, (short)0, c, false, false);
}

// ---------------- TDM: 2D f16 tile -> LDS (CDNA5 ISA) ------
// tile_rows x tile_cols f16 elements (row-major, row stride = stride elems)
// -> LDS at lds_off, 16B pad appended per pad-interval (code: 3=16DW, 4=32DW).
DEV void tdm_load_2d(u32 lds_off, const f16* gptr,
                     u32 tensor_rows, u32 tensor_cols, u32 stride,
                     u32 tile_cols, u32 tile_rows, u32 pad_code) {
    u64 ga = (u64)(uintptr_t)gptr;
    u32x4 g0;
    g0[0] = 1u;                                        // count=1, user mode
    g0[1] = lds_off;                                   // lds_addr (bytes)
    g0[2] = (u32)ga;                                   // global_addr[31:0]
    g0[3] = (u32)((ga >> 32) & 0x01FFFFFFu) | (2u << 30); // ga[56:32] | type=2
    i32x8 g1;
    g1[0] = (int)((1u << 16)          // data_size = 2 bytes
                | (1u << 20)          // pad_enable
                | (pad_code << 22)    // pad_interval
                | (3u << 25));        // pad_amount code 3 = 4 DWORDs (16B)
    g1[1] = (int)((tensor_cols & 0xFFFFu) << 16);      // tensor_dim0[15:0]
    g1[2] = (int)(((tensor_cols >> 16) & 0xFFFFu)
                | ((tensor_rows & 0xFFFFu) << 16));    // dim0 hi | dim1 lo
    g1[3] = (int)(((tensor_rows >> 16) & 0xFFFFu)
                | ((tile_cols & 0xFFFFu) << 16));      // dim1 hi | tile_dim0
    g1[4] = (int)(tile_rows & 0xFFFFu);                // tile_dim1 (tile_dim2=0)
    g1[5] = (int)stride;                               // tensor_dim0_stride lo
    g1[6] = 0;
    g1[7] = 0;
    i32x4 z4 = {0, 0, 0, 0};
    i32x8 z8 = {0, 0, 0, 0, 0, 0, 0, 0};
    __builtin_amdgcn_tensor_load_to_lds(g0, g1, z4, z4, z8, 0);
}

// ---------------- fp32 -> f16 convert --------------------------------------
__global__ void cvt_f32_f16_kernel(const float* __restrict__ in,
                                   f16* __restrict__ out, int n) {
    int i = blockIdx.x * blockDim.x + threadIdx.x;
    if (i < n) out[i] = (f16)in[i];
}

// ---------------- TDM double-buffered GEMM ---------------------------------
// C[m,n] = sum_c A[m,c]*B[n,c]; A [MTOT,1024], B [brows,1024] f16 row-major.
// Block tile 128x128, 8 waves (4x2), wave tile 32x64.  Wave 0 drives the TDM.
// MODE 0: scatter to q/k row-major [B,H,T,64] and v transposed [B,H,64,T].
// MODE 1: store fp32 C [MTOT,1024].
template <int MODE>
__global__ void __launch_bounds__(256)
gemm_tdm_kernel(const f16* __restrict__ A, const f16* __restrict__ B, int brows,
                f16* __restrict__ qb, f16* __restrict__ kb, f16* __restrict__ vtb,
                float* __restrict__ Cout) {
    const int K = DIM;
    __shared__ __align__(128) char smA[2][128 * PITCH];
    __shared__ __align__(128) char smB[2][128 * PITCH];

    int lane = threadIdx.x & 31;
    int wave = threadIdx.x >> 5;
    int lh = lane >> 4, ln = lane & 15;
    int wm = wave & 3, wn = wave >> 2;
    int blockM = blockIdx.x * 128;
    int blockN = blockIdx.y * 128;
    int m0 = wm * 32;                  // wave tile origin inside block tile
    int n0 = wn * 64;

    v8f acc[2][4];
#pragma unroll
    for (int mt = 0; mt < 2; ++mt)
#pragma unroll
        for (int nt = 0; nt < 4; ++nt) acc[mt][nt] = {};

    u32 offA[2] = { (u32)(uintptr_t)&smA[0][0], (u32)(uintptr_t)&smA[1][0] };
    u32 offB[2] = { (u32)(uintptr_t)&smB[0][0], (u32)(uintptr_t)&smB[1][0] };
    const f16* gA = A + (size_t)blockM * K;
    const f16* gB = B + (size_t)blockN * K;

    const int NK = K / KSTEP;          // 32
    if (wave == 0) {
        tdm_load_2d(offA[0], gA,         MTOT,       K, K, KSTEP, 128, 3);
        tdm_load_2d(offB[0], gB,         (u32)brows, K, K, KSTEP, 128, 3);
        tdm_load_2d(offA[1], gA + KSTEP, MTOT,       K, K, KSTEP, 128, 3);
        tdm_load_2d(offB[1], gB + KSTEP, (u32)brows, K, K, KSTEP, 128, 3);
    }

    for (int kk = 0; kk < NK; ++kk) {
        int cur = kk & 1;
        if (wave == 0) {
            if (kk + 1 < NK) __builtin_amdgcn_s_wait_tensorcnt(2);
            else             __builtin_amdgcn_s_wait_tensorcnt(0);
        }
        __syncthreads();               // tiles `kk` visible to all waves

        const char* tA = smA[cur];
        const char* tB = smB[cur];
        v16h af[2], bf[4];
#pragma unroll
        for (int mt = 0; mt < 2; ++mt)
            af[mt] = lds_frag_a(tA + (m0 + mt * 16 + ln) * PITCH, lh);
#pragma unroll
        for (int nt = 0; nt < 4; ++nt)
            bf[nt] = lds_frag_b(tB + (n0 + nt * 16 + ln) * PITCH, lh);
#pragma unroll
        for (int mt = 0; mt < 2; ++mt)
#pragma unroll
            for (int nt = 0; nt < 4; ++nt)
                acc[mt][nt] = wmma_f16(af[mt], bf[nt], acc[mt][nt]);

        __syncthreads();               // everyone done reading buffer `cur`
        if (wave == 0 && kk + 2 < NK) {
            int kn = (kk + 2) * KSTEP;
            tdm_load_2d(offA[cur], gA + kn, MTOT,       K, K, KSTEP, 128, 3);
            tdm_load_2d(offB[cur], gB + kn, (u32)brows, K, K, KSTEP, 128, 3);
        }
    }

#pragma unroll
    for (int mt = 0; mt < 2; ++mt)
#pragma unroll
        for (int nt = 0; nt < 4; ++nt)
#pragma unroll
            for (int v = 0; v < 8; ++v) {
                int gm = blockM + m0 + mt * 16 + v + lh * 8;
                int gn = blockN + n0 + nt * 16 + ln;
                if (MODE == 1) {
                    Cout[(size_t)gm * DIM + gn] = acc[mt][nt][v];
                } else {
                    f16 val = (f16)acc[mt][nt][v];
                    int b = gm >> 11, t = gm & (SEQ - 1);
                    int sel = gn >> 10, c = gn & (DIM - 1);
                    int h = c >> 6, d = c & 63;
                    size_t bh = (size_t)(b * NHEADS + h);
                    if (sel == 0)      qb[(bh * SEQ + t) * HDIM + d] = val;
                    else if (sel == 1) kb[(bh * SEQ + t) * HDIM + d] = val;
                    else               vtb[(bh * HDIM + d) * SEQ + t] = val;
                }
            }
}

// ---------------- flash attention, TDM-staged K/V^T ------------------------
// One block = (b,h) x 128 queries; one wave = 16 queries.  Wave 0 DMAs the
// shared 32-key K tile (32x64) and V^T tile (64x32) into LDS, double-buffered.
// S^T = Kmat * Q^T so each lane owns one query column of the C fragment;
// softmax stats = per-lane reductions + one cross-half shfl_xor(16).
// Then O^T = V^T * P^T (P^T as B operand, built from the S^T C fragments).
__global__ void __launch_bounds__(256)
attn_kernel(const f16* __restrict__ q, const f16* __restrict__ k,
            const f16* __restrict__ vt, f16* __restrict__ o) {
    const int T = SEQ, D = HDIM;
    __shared__ __align__(128) char smK[2][32 * PITCHK];  // 32 keys x 64 c
    __shared__ __align__(128) char smV[2][64 * PITCH];   // 64 d x 32 keys

    int lane = threadIdx.x & 31;
    int wave = threadIdx.x >> 5;
    int lh = lane >> 4, ln = lane & 15;
    int bh   = blockIdx.x >> 4;        // 0..63 = b*16+h
    int qblk = blockIdx.x & 15;        // 16 query-blocks of 128
    int q0 = qblk * 128 + wave * 16;

    const f16* Q  = q  + (size_t)bh * T * D;
    const f16* Kp = k  + (size_t)bh * T * D;
    const f16* VT = vt + (size_t)bh * D * T;

    v16h bq0 = load_frag_b(Q + (size_t)(q0 + ln) * D + 0  + lh * 8); // c 0..31
    v16h bq1 = load_frag_b(Q + (size_t)(q0 + ln) * D + 32 + lh * 8); // c 32..63

    u32 offK[2] = { (u32)(uintptr_t)&smK[0][0], (u32)(uintptr_t)&smK[1][0] };
    u32 offV[2] = { (u32)(uintptr_t)&smV[0][0], (u32)(uintptr_t)&smV[1][0] };

    v8f acc0 = {}, acc1 = {}, acc2 = {}, acc3 = {};    // O^T, d-tiles 0..3
    float run_max = -__builtin_inff();
    float run_sum = 0.0f;
    const float scale = 0.125f;                        // 1/sqrt(64)

    const int NJ = T / 32;             // 64 key-chunks
    if (wave == 0) {
        tdm_load_2d(offK[0], Kp,           T, D, D, 64, 32, 4);
        tdm_load_2d(offV[0], VT,           D, T, T, 32, 64, 3);
        tdm_load_2d(offK[1], Kp + 32 * D,  T, D, D, 64, 32, 4);
        tdm_load_2d(offV[1], VT + 32,      D, T, T, 32, 64, 3);
    }

    for (int jj = 0; jj < NJ; ++jj) {
        int cur = jj & 1;
        if (wave == 0) {
            if (jj + 1 < NJ) __builtin_amdgcn_s_wait_tensorcnt(2);
            else             __builtin_amdgcn_s_wait_tensorcnt(0);
        }
        __syncthreads();               // K/V^T chunk `jj` visible

        const char* tK = smK[cur];
        const char* tV = smV[cur];

        // S^T tiles: A rows = keys (local rows of K tile), c groups 0 / 32
        v8f s0 = {}, s1 = {};
        s0 = wmma_f16(lds_frag_a(tK + ln * PITCHK,              lh), bq0, s0);
        s0 = wmma_f16(lds_frag_a(tK + ln * PITCHK + 64,         lh), bq1, s0);
        s1 = wmma_f16(lds_frag_a(tK + (16 + ln) * PITCHK,       lh), bq0, s1);
        s1 = wmma_f16(lds_frag_a(tK + (16 + ln) * PITCHK + 64,  lh), bq1, s1);

        float tmax = -__builtin_inff();
#pragma unroll
        for (int v = 0; v < 8; ++v) {
            s0[v] *= scale; s1[v] *= scale;
            tmax = fmaxf(tmax, fmaxf(s0[v], s1[v]));
        }
        tmax = fmaxf(tmax, __shfl_xor(tmax, 16));      // other 16 keys, same query
        float new_max = fmaxf(run_max, tmax);
        float alpha = __expf(run_max - new_max);

        float p0[8], p1[8];
        float psum = 0.0f;
#pragma unroll
        for (int v = 0; v < 8; ++v) {
            p0[v] = __expf(s0[v] - new_max);
            p1[v] = __expf(s1[v] - new_max);
            psum += p0[v] + p1[v];
        }
        psum += __shfl_xor(psum, 16);
        run_sum = run_sum * alpha + psum;
        run_max = new_max;

#pragma unroll
        for (int v = 0; v < 8; ++v) {
            acc0[v] *= alpha; acc1[v] *= alpha; acc2[v] *= alpha; acc3[v] *= alpha;
        }

        // P^T as B operand for K-dim 32 (keys j0..j0+31)
        v16h bp;
#pragma unroll
        for (int v = 0; v < 8; ++v) {
            float o0 = __shfl_xor(p0[v], 16);
            float o1 = __shfl_xor(p1[v], 16);
            bp[v]     = (f16)p0[v];
            bp[v + 8] = (f16)p1[v];
        }

        // O^T += V^T * P^T ; A rows = head dims from the V^T LDS tile
        acc0 = wmma_f16(lds_frag_a(tV + (ln)      * PITCH, lh), bp, acc0);
        acc1 = wmma_f16(lds_frag_a(tV + (16 + ln) * PITCH, lh), bp, acc1);
        acc2 = wmma_f16(lds_frag_a(tV + (32 + ln) * PITCH, lh), bp, acc2);
        acc3 = wmma_f16(lds_frag_a(tV + (48 + ln) * PITCH, lh), bp, acc3);

        __syncthreads();               // done reading buffer `cur`
        if (wave == 0 && jj + 2 < NJ) {
            int jn = (jj + 2) * 32;
            tdm_load_2d(offK[cur], Kp + (size_t)jn * D, T, D, D, 64, 32, 4);
            tdm_load_2d(offV[cur], VT + jn,             D, T, T, 32, 64, 3);
        }
    }

    float inv = 1.0f / run_sum;
    int b = bh >> 4, h = bh & 15;
    int tq = q0 + ln;
    f16* ob = o + ((size_t)b * T + tq) * DIM + h * HDIM;
#pragma unroll
    for (int v = 0; v < 8; ++v) {
        int d = v + lh * 8;
        ob[d]      = (f16)(acc0[v] * inv);
        ob[16 + d] = (f16)(acc1[v] * inv);
        ob[32 + d] = (f16)(acc2[v] * inv);
        ob[48 + d] = (f16)(acc3[v] * inv);
    }
}

// ---------------- host-side launch -----------------------------------------
extern "C" void kernel_launch(void* const* d_in, const int* in_sizes, int n_in,
                              void* d_out, int out_size, void* d_ws, size_t ws_size,
                              hipStream_t stream) {
    const float* x     = (const float*)d_in[0];   // [4,2048,1024]
    const float* w_qkv = (const float*)d_in[1];   // [3072,1024]
    const float* w_out = (const float*)d_in[2];   // [1024,1024]
    float* out = (float*)d_out;                   // [4,2048,1024] fp32

    const size_t NX    = (size_t)MTOT * DIM;
    const size_t NWQKV = (size_t)3 * DIM * DIM;
    const size_t NWOUT = (size_t)DIM * DIM;
    const size_t NHEADBUF = (size_t)BATCH * NHEADS * SEQ * HDIM;

    char* ws = (char*)d_ws;
    f16* xb    = (f16*)ws;                          ws += NX * 2;
    f16* wqkvb = (f16*)ws;                          ws += NWQKV * 2;
    f16* woutb = (f16*)ws;                          ws += NWOUT * 2;
    f16* qb    = (f16*)ws;                          ws += NHEADBUF * 2;
    f16* kb    = (f16*)ws;                          ws += NHEADBUF * 2;
    f16* vtb   = (f16*)ws;                          ws += NHEADBUF * 2;
    f16* ab    = (f16*)ws;                          ws += NX * 2;

    cvt_f32_f16_kernel<<<(int)((NX    + 255) / 256), 256, 0, stream>>>(x,     xb,    (int)NX);
    cvt_f32_f16_kernel<<<(int)((NWQKV + 255) / 256), 256, 0, stream>>>(w_qkv, wqkvb, (int)NWQKV);
    cvt_f32_f16_kernel<<<(int)((NWOUT + 255) / 256), 256, 0, stream>>>(w_out, woutb, (int)NWOUT);

    gemm_tdm_kernel<0><<<dim3(MTOT / 128, (3 * DIM) / 128), 256, 0, stream>>>(
        xb, wqkvb, 3 * DIM, qb, kb, vtb, nullptr);

    attn_kernel<<<BATCH * NHEADS * (SEQ / 128), 256, 0, stream>>>(qb, kb, vtb, ab);

    gemm_tdm_kernel<1><<<dim3(MTOT / 128, DIM / 128), 256, 0, stream>>>(
        ab, woutb, DIM, nullptr, nullptr, nullptr, out);
}
